// NemoRGCN_11458972746385
// MI455X (gfx1250) — matrix operand order repaired and checked by reading, not compiled, not run on hardware
//
#include <hip/hip_runtime.h>
#include <hip/hip_bf16.h>

// ---------------------------------------------------------------------------
// RGCN (2 layers, mean aggregation, 8 relations) + linear classifier.
// Strategy: aggregate-then-transform. Per relation r:
//   agg_r[dst] = sum_{e: type r -> dst} x[src]          (float atomics, L2)
//   out = sum_r (diag(1/cnt_r) * agg_r) @ W_r + x @ root + b
// GEMMs run in bf16 WMMA (v_wmma_f32_16x16x32_bf16, wave32), f32 accumulate.
// ---------------------------------------------------------------------------

typedef float    v8f   __attribute__((ext_vector_type(8)));
typedef __bf16   v16bf __attribute__((ext_vector_type(16)));
typedef __bf16   v8bf  __attribute__((ext_vector_type(8)));

#define RREL   8
#define NMAT   9        // 8 relations + root
#define DIM    128
#define TPAD   136      // LDS A-tile row pitch in bf16: 68 dwords -> bank stride 4

// ------------------------- small helper kernels ----------------------------

__global__ __launch_bounds__(256)
void k_count_edges(const int* __restrict__ edst, const int* __restrict__ etype,
                   int* __restrict__ cnt, int E, int N) {
    int e = blockIdx.x * 256 + threadIdx.x;
    if (e < E) atomicAdd(&cnt[etype[e] * N + edst[e]], 1);
}

__global__ __launch_bounds__(256)
void k_inv(const int* __restrict__ cnt, float* __restrict__ inv, int n) {
    int i = blockIdx.x * 256 + threadIdx.x;
    if (i < n) {
        int c = cnt[i];
        inv[i] = c > 0 ? 1.0f / (float)c : 0.0f;
    }
}

// dst[(m*cols + o)*rows + i] = (bf16) src[(m*rows + i)*cols + o]
__global__ __launch_bounds__(256)
void k_cvtT(const float* __restrict__ src, __bf16* __restrict__ dst,
            int rows, int cols, int mats) {
    int idx = blockIdx.x * 256 + threadIdx.x;
    int per = rows * cols;
    if (idx < mats * per) {
        int m = idx / per, rem = idx - m * per;
        int i = rem / cols, o = rem - i * cols;
        dst[((size_t)m * cols + o) * rows + i] = (__bf16)src[idx];
    }
}

// One wave per edge: 32 lanes x float4 = 128 floats; atomic add into agg.
__global__ __launch_bounds__(256)
void k_scatter(const float* __restrict__ feat, const int* __restrict__ esrc,
               const int* __restrict__ edst, const int* __restrict__ etype,
               float* __restrict__ agg, int E, int N) {
    int lane  = threadIdx.x & 31;
    int gwave = (blockIdx.x * 256 + threadIdx.x) >> 5;
    int nwave = gridDim.x * 8;
    for (int e = gwave; e < E; e += nwave) {
        int s = esrc[e], d = edst[e], t = etype[e];
        const float4 v = *(const float4*)(feat + (size_t)s * DIM + lane * 4);
        float* p = agg + ((size_t)t * N + d) * DIM + (size_t)lane * 4;
        atomicAdd(p + 0, v.x);
        atomicAdd(p + 1, v.y);
        atomicAdd(p + 2, v.z);
        atomicAdd(p + 3, v.w);
    }
}

// ------------------------- fused RGCN layer (WMMA) -------------------------
// Block = 256 threads = 8 waves. Block handles one 16-row node tile; wave w
// owns output columns [16w, 16w+16). A tile (16x128) is staged in LDS with
// the 1/cnt scaling and bf16 conversion applied once, then each wave runs
// 4 K-steps of v_wmma_f32_16x16x32_bf16 per matrix (9 matrices, 36 WMMAs).
__global__ __launch_bounds__(256)
void k_rgcn_layer(const float* __restrict__ xin,        // [N,128]
                  const float* __restrict__ agg,        // [R,N,128]
                  const float* __restrict__ invc,       // [R*N]
                  const __bf16* __restrict__ Wt,        // [NMAT,128(o),128(k)]
                  const float* __restrict__ bias,       // [128]
                  float* __restrict__ out,              // [N,128]
                  int N, int relu) {
    __shared__ __bf16 As[16 * TPAD];

    const int tid  = threadIdx.x;
    const int lane = tid & 31;
    const int wave = tid >> 5;              // column tile id, 0..7
    const int ntile = N >> 4;               // 50000/16 = 3125 exactly

    for (int tile = blockIdx.x; tile < ntile; tile += gridDim.x) {
        const int rowbase = tile * 16;
        v8f acc = {};

        for (int m = 0; m < NMAT; ++m) {
            // ---- stage scaled bf16 A tile: 256 thr x 8 elems = 16x128 ----
            {
                const int r  = tid >> 4;            // 0..15
                const int c0 = (tid & 15) * 8;      // 0..120
                const float* src = (m < RREL)
                    ? agg + (((size_t)m * N + rowbase + r) * DIM + c0)
                    : xin + ((size_t)(rowbase + r) * DIM + c0);
                const float s = (m < RREL) ? invc[m * N + rowbase + r] : 1.0f;
                float4 f0 = *(const float4*)(src);
                float4 f1 = *(const float4*)(src + 4);
                __bf16* d = &As[r * TPAD + c0];
                d[0] = (__bf16)(f0.x * s); d[1] = (__bf16)(f0.y * s);
                d[2] = (__bf16)(f0.z * s); d[3] = (__bf16)(f0.w * s);
                d[4] = (__bf16)(f1.x * s); d[5] = (__bf16)(f1.y * s);
                d[6] = (__bf16)(f1.z * s); d[7] = (__bf16)(f1.w * s);
            }
            __syncthreads();

            // weight rows for this wave's 16 output columns (L2-hot, 288KB)
            const __bf16* wbase =
                Wt + ((size_t)m * DIM + wave * 16 + (lane & 15)) * DIM;
            if (m + 1 < NMAT)   // hint next matrix's weight slice into cache
                __builtin_prefetch(wbase + (size_t)DIM * DIM, 0, 3);

            const int arow = lane & 15;
            #pragma unroll
            for (int ks = 0; ks < 4; ++ks) {
                // A fragment: lane<16 -> K {k..k+15}; lane>=16 -> K {k+16..k+31}
                const int k0 = ks * 32 + ((lane & 16) ? 16 : 0);
                const __bf16* ap = &As[arow * TPAD + k0];
                const v8bf lo = *(const v8bf*)(ap);
                const v8bf hi = *(const v8bf*)(ap + 8);
                v16bf a;
                #pragma unroll
                for (int i = 0; i < 8; ++i) { a[i] = lo[i]; a[8 + i] = hi[i]; }
                // B fragment: contiguous 16 bf16 of Wt row (pre-transposed)
                const __bf16* bp = wbase + ks * 32 + ((lane & 16) ? 16 : 0);
                const v16bf b = *(const v16bf*)(bp);
                acc = __builtin_amdgcn_wmma_f32_16x16x32_bf16(
                    false, a, false, b, (short)0, acc, false, false);
            }
            __syncthreads();
        }

        // ---- epilogue: bias (+ReLU), C layout: vgpr j -> M=j(+8), N=lane&15
        const int c  = lane & 15;
        const float bz = bias[wave * 16 + c];
        #pragma unroll
        for (int j = 0; j < 8; ++j) {
            const int rr = j + ((lane & 16) ? 8 : 0);
            float v = acc[j] + bz;
            if (relu) v = fmaxf(v, 0.0f);
            out[(size_t)(rowbase + rr) * DIM + wave * 16 + c] = v;
        }
    }
}

// ------------------------- classifier (WMMA, C=16) --------------------------
__global__ __launch_bounds__(256)
void k_classifier(const float* __restrict__ emb,      // [N,128]
                  const __bf16* __restrict__ Wtc,     // [16(o),128(k)]
                  const float* __restrict__ bc,       // [16]
                  float* __restrict__ out,            // [N,16]
                  int N) {
    const int lane = threadIdx.x & 31;
    const int wave = threadIdx.x >> 5;
    const int ntile = N >> 4;
    for (int tile = blockIdx.x * 8 + wave; tile < ntile; tile += gridDim.x * 8) {
        const int rowbase = tile * 16;
        v8f acc = {};
        const float*  Arow  = emb + (size_t)(rowbase + (lane & 15)) * DIM;
        const __bf16* wbase = Wtc + (size_t)(lane & 15) * DIM;
        #pragma unroll
        for (int ks = 0; ks < 4; ++ks) {
            const int k0 = ks * 32 + ((lane & 16) ? 16 : 0);
            float4 f0 = *(const float4*)(Arow + k0);
            float4 f1 = *(const float4*)(Arow + k0 + 4);
            float4 f2 = *(const float4*)(Arow + k0 + 8);
            float4 f3 = *(const float4*)(Arow + k0 + 12);
            v16bf a;
            a[0]=(__bf16)f0.x; a[1]=(__bf16)f0.y; a[2]=(__bf16)f0.z; a[3]=(__bf16)f0.w;
            a[4]=(__bf16)f1.x; a[5]=(__bf16)f1.y; a[6]=(__bf16)f1.z; a[7]=(__bf16)f1.w;
            a[8]=(__bf16)f2.x; a[9]=(__bf16)f2.y; a[10]=(__bf16)f2.z; a[11]=(__bf16)f2.w;
            a[12]=(__bf16)f3.x; a[13]=(__bf16)f3.y; a[14]=(__bf16)f3.z; a[15]=(__bf16)f3.w;
            const v16bf b = *(const v16bf*)(wbase + ks * 32 + ((lane & 16) ? 16 : 0));
            acc = __builtin_amdgcn_wmma_f32_16x16x32_bf16(
                false, a, false, b, (short)0, acc, false, false);
        }
        const int c = lane & 15;
        const float bz = bc[c];
        #pragma unroll
        for (int j = 0; j < 8; ++j) {
            const int rr = j + ((lane & 16) ? 8 : 0);
            out[(size_t)(rowbase + rr) * 16 + c] = acc[j] + bz;
        }
    }
}

// ---------------------------------------------------------------------------

extern "C" void kernel_launch(void* const* d_in, const int* in_sizes, int n_in,
                              void* d_out, int out_size, void* d_ws, size_t ws_size,
                              hipStream_t stream) {
    const float* x     = (const float*)d_in[0];
    const int*   eidx  = (const int*)d_in[1];   // [2,E]
    const int*   etype = (const int*)d_in[2];
    const float* W1    = (const float*)d_in[3];
    const float* root1 = (const float*)d_in[4];
    const float* b1    = (const float*)d_in[5];
    const float* W2    = (const float*)d_in[6];
    const float* root2 = (const float*)d_in[7];
    const float* b2    = (const float*)d_in[8];
    const float* Wc    = (const float*)d_in[9];
    const float* bc    = (const float*)d_in[10];

    const int N = in_sizes[0] / DIM;            // 50000 (multiple of 16)
    const int E = in_sizes[2];                  // 600000
    const int* esrc = eidx;
    const int* edst = eidx + E;

    // ---- workspace carve-up -------------------------------------------------
    char* ws = (char*)d_ws;
    size_t off = 0;
    auto take = [&](size_t bytes) {
        char* p = ws + off;
        off = (off + bytes + 255) & ~(size_t)255;
        return p;
    };
    int*    cnt  = (int*)   take((size_t)RREL * N * sizeof(int));
    float*  invc = (float*) take((size_t)RREL * N * sizeof(float));
    float*  agg  = (float*) take((size_t)RREL * N * DIM * sizeof(float));
    float*  h    = (float*) take((size_t)N * DIM * sizeof(float));
    __bf16* Wt1  = (__bf16*)take((size_t)NMAT * DIM * DIM * sizeof(__bf16));
    __bf16* Wt2  = (__bf16*)take((size_t)NMAT * DIM * DIM * sizeof(__bf16));
    __bf16* Wtc  = (__bf16*)take((size_t)16 * DIM * sizeof(__bf16));
    (void)ws_size;

    float* out = (float*)d_out;                 // [N,16]
    float* emb = (float*)d_out + (size_t)N * 16; // [N,128]

    const int ntile = N >> 4;                   // 3125

    // ---- 1. edge counts -> 1/cnt (shared by both layers) --------------------
    hipMemsetAsync(cnt, 0, (size_t)RREL * N * sizeof(int), stream);
    k_count_edges<<<(E + 255) / 256, 256, 0, stream>>>(edst, etype, cnt, E, N);
    k_inv<<<(RREL * N + 255) / 256, 256, 0, stream>>>(cnt, invc, RREL * N);

    // ---- 2. weights -> bf16, pre-transposed [mat][o][k] ---------------------
    {
        int per = DIM * DIM;
        k_cvtT<<<(RREL * per + 255) / 256, 256, 0, stream>>>(W1, Wt1, DIM, DIM, RREL);
        k_cvtT<<<(per + 255) / 256, 256, 0, stream>>>(root1, Wt1 + (size_t)RREL * per, DIM, DIM, 1);
        k_cvtT<<<(RREL * per + 255) / 256, 256, 0, stream>>>(W2, Wt2, DIM, DIM, RREL);
        k_cvtT<<<(per + 255) / 256, 256, 0, stream>>>(root2, Wt2 + (size_t)RREL * per, DIM, DIM, 1);
        k_cvtT<<<(DIM * 16 + 255) / 256, 256, 0, stream>>>(Wc, Wtc, DIM, 16, 1);
    }

    // ---- 3. layer 1: scatter x -> agg, fused GEMM + ReLU --------------------
    hipMemsetAsync(agg, 0, (size_t)RREL * N * DIM * sizeof(float), stream);
    k_scatter<<<2048, 256, 0, stream>>>(x, esrc, edst, etype, agg, E, N);
    k_rgcn_layer<<<ntile, 256, 0, stream>>>(x, agg, invc, Wt1, b1, h, N, 1);

    // ---- 4. layer 2: scatter h -> agg, fused GEMM (emb straight to d_out) ---
    hipMemsetAsync(agg, 0, (size_t)RREL * N * DIM * sizeof(float), stream);
    k_scatter<<<2048, 256, 0, stream>>>(h, esrc, edst, etype, agg, E, N);
    k_rgcn_layer<<<ntile, 256, 0, stream>>>(h, agg, invc, Wt2, b2, emb, N, 0);

    // ---- 5. classifier ------------------------------------------------------
    k_classifier<<<(ntile + 7) / 8, 256, 0, stream>>>(emb, Wtc, bc, out, N);
}
